// istsplm_forecast_77567109366119
// MI455X (gfx1250) — matrix operand order repaired and checked by reading, not compiled, not run on hardware
//
#include <hip/hip_runtime.h>
#include <hip/hip_bf16.h>

typedef _Float16 f16;
typedef __attribute__((ext_vector_type(16))) _Float16 v16h;
typedef __attribute__((ext_vector_type(8)))  _Float16 v8h;
typedef __attribute__((ext_vector_type(8)))  float    v8f;

#define S_LEN  4096
#define DMODEL 1024
#define NH     16
#define HDIM   64

#define XT_STRIDE 88    // f16 units; 176B rows: 16B aligned, conflict-free A reads
#define WT_STRIDE 136   // f16 units; 272B rows: 16B aligned, 2-way on B reads
#define P_STRIDE  56    // f16 units; 112B rows: 16B aligned, conflict-free

#define XT_BUF (32 * XT_STRIDE)
#define WT_BUF (64 * WT_STRIDE)

static __device__ __forceinline__ v8f wmma_f16(v16h a, v16h b, v8f c) {
  return __builtin_amdgcn_wmma_f32_16x16x32_f16(false, a, false, b, (short)0, c,
                                                false, false);
}

union V16U { v16h v; v8h h[2]; };

// CDNA5 async copy: global -> LDS, 16B per lane, tracked by ASYNCcnt.
// LDS operand is the wave-relative byte offset (= low 32 bits of the generic
// pointer for LDS on gfx12).
static __device__ __forceinline__ void async_ld_b128(const void* gbase,
                                                     unsigned goff_bytes,
                                                     void* lds_dst) {
  unsigned l = (unsigned)(uintptr_t)lds_dst;
  asm volatile("global_load_async_to_lds_b128 %0, %1, %2"
               :: "v"(l), "v"(goff_bytes), "s"(gbase)
               : "memory");
}

// ---------------------------------------------------------------------------
// Pack pass 1: hidden f32 -> f16 (vectorized elementwise)
// ---------------------------------------------------------------------------
__global__ __launch_bounds__(256) void convert_x_kernel(
    const float* __restrict__ X, f16* __restrict__ Xf) {
  const size_t i = ((size_t)blockIdx.x * 256 + threadIdx.x) * 8;
  v8h t;
#pragma unroll
  for (int j = 0; j < 8; ++j) t[j] = (f16)X[i + j];
  *(v8h*)(Xf + i) = t;
}

// ---------------------------------------------------------------------------
// Pack pass 2: W[o][d] f32 -> WT[d][o] f16 (LDS-tiled transpose, one-time)
// ---------------------------------------------------------------------------
__global__ __launch_bounds__(256) void transpose_w_kernel(
    const float* __restrict__ Wq, const float* __restrict__ Wk,
    const float* __restrict__ Wv, const float* __restrict__ Wo,
    f16* __restrict__ WqT, f16* __restrict__ WkT,
    f16* __restrict__ WvT, f16* __restrict__ WoT) {
  __shared__ float t[32][33];
  const int z = blockIdx.z;
  const float* W = (z == 0) ? Wq : (z == 1) ? Wk : (z == 2) ? Wv : Wo;
  f16* WT       = (z == 0) ? WqT : (z == 1) ? WkT : (z == 2) ? WvT : WoT;
  const int o0 = blockIdx.x * 32, d0 = blockIdx.y * 32;
  const int c = threadIdx.x & 31, r0 = threadIdx.x >> 5;
#pragma unroll
  for (int i = 0; i < 4; ++i) {
    const int r = r0 + i * 8;
    t[r][c] = W[(size_t)(o0 + r) * DMODEL + d0 + c];
  }
  __syncthreads();
#pragma unroll
  for (int i = 0; i < 4; ++i) {
    const int r = r0 + i * 8;
    WT[(size_t)(d0 + r) * DMODEL + o0 + c] = (f16)t[c][r];
  }
}

// ---------------------------------------------------------------------------
// Kernel 1: fused QKV projection (+bias) + continuous-time RoPE + f16 pack.
// Block: 256 thr = 8 waves; tile 32 (M) x 128 (N = 2 heads). Double-buffered
// LDS stages filled with global_load_async_to_lds_b128 (5 chunks/thread/stage).
// Each wave owns cols (base, base+32) so RoPE pairs stay in-register.
// ---------------------------------------------------------------------------
__global__ __launch_bounds__(256) void qkv_rope_kernel(
    const f16* __restrict__ Xf, const float* __restrict__ TS,
    const f16* __restrict__ WqT, const float* __restrict__ bq,
    const f16* __restrict__ WkT, const float* __restrict__ bk,
    const f16* __restrict__ WvT, const float* __restrict__ bv,
    f16* __restrict__ Qh, f16* __restrict__ KT, f16* __restrict__ Vh) {
  __shared__ __align__(16) f16 Xt[2 * XT_BUF];
  __shared__ __align__(16) f16 Wt[2 * WT_BUF];

  const int z = blockIdx.z;
  const f16*   WT = (z == 0) ? WqT : (z == 1) ? WkT : WvT;
  const float* bb = (z == 0) ? bq : (z == 1) ? bk : bv;

  const int m0  = blockIdx.x * 32;
  const int n0  = blockIdx.y * 128;
  const int tid = threadIdx.x;
  const int lane = tid & 31;
  const int w = tid >> 5;
  const int mi = w >> 2;          // 0/1: 16-row half
  const int j  = w & 3;           // 0..3
  const int headq = j >> 1;       // head within block (2 heads / 128 cols)
  const int jj = j & 1;           // 16-col tile within first half of head
  const int l15 = lane & 15;
  const int lhalf = lane >> 4;

  // per-thread chunk coordinates for the async stage fills
  const int xr = tid >> 3, xc8 = (tid & 7) << 3;          // X: 32 rows x 8 chunks

  v8f acc_lo = {};
  v8f acc_hi = {};

  // issue stage kc into LDS buffer `buf` (1 X chunk + 4 W chunks per thread)
  auto issue = [&](int kc, int buf) {
    async_ld_b128(Xf, (unsigned)(((m0 + xr) * DMODEL + kc + xc8) * 2),
                  &Xt[buf * XT_BUF + xr * XT_STRIDE + xc8]);
#pragma unroll
    for (int it = 0; it < 4; ++it) {
      const int idx = tid + it * 256;            // 1024 chunks: 64 k x 16
      const int k = idx >> 4, c8 = (idx & 15) << 3;
      async_ld_b128(WT, (unsigned)(((kc + k) * DMODEL + n0 + c8) * 2),
                    &Wt[buf * WT_BUF + k * WT_STRIDE + c8]);
    }
  };

  issue(0, 0);
  for (int st = 0; st < DMODEL / 64; ++st) {
    if (st + 1 < DMODEL / 64) {
      issue((st + 1) * 64, (st + 1) & 1);
      asm volatile("s_wait_asynccnt 0x5" ::: "memory");  // stage st complete
    } else {
      asm volatile("s_wait_asynccnt 0x0" ::: "memory");
    }
    __syncthreads();

    const f16* xb = &Xt[(st & 1) * XT_BUF];
    const f16* wb = &Wt[(st & 1) * WT_BUF];
#pragma unroll
    for (int c = 0; c < 2; ++c) {
      V16U a;  // A frag: lane row = l15, K split per lane half
      const int abase = (mi * 16 + l15) * XT_STRIDE + c * 32 + lhalf * 8;
      a.h[0] = *(const v8h*)(&xb[abase]);
      a.h[1] = *(const v8h*)(&xb[abase + 16]);

      V16U blo, bhi;  // B frag: lane col = l15, K split per lane half
      const int nlo = headq * 64 + jj * 16 + l15;
#pragma unroll
      for (int s = 0; s < 16; ++s) {
        const int kb = (c * 32 + lhalf * 8 + (s & 7) + (s >> 3) * 16) * WT_STRIDE;
        blo.v[s] = wb[kb + nlo];
        bhi.v[s] = wb[kb + nlo + 32];
      }

      acc_lo = wmma_f16(a.v, blo.v, acc_lo);
      acc_hi = wmma_f16(a.v, bhi.v, acc_hi);
    }
    __syncthreads();
  }

  // epilogue: bias, RoPE (q,k), 1/sqrt(HD) folded into q, pack f16
  const int head  = (n0 >> 6) + headq;   // global head index
  const int hd_lo = jj * 16 + l15;       // 0..31
  const float b_lo = bb[n0 + headq * 64 + hd_lo];
  const float b_hi = bb[n0 + headq * 64 + hd_lo + 32];
  // inv_freq = theta^(-hd/32)
  const float invf = __expf(-logf(10000.0f) * (float)hd_lo * (1.0f / 32.0f));

#pragma unroll
  for (int v = 0; v < 8; ++v) {
    const int srow = m0 + mi * 16 + v + 8 * lhalf;
    float lo = acc_lo[v] + b_lo;
    float hi = acc_hi[v] + b_hi;
    if (z < 2) {
      const float ang = TS[srow] * invf;
      float sn, cs;
      __sincosf(ang, &sn, &cs);
      const float nlo = lo * cs - hi * sn;   // rotate_half: [-x2, x1]
      const float nhi = hi * cs + lo * sn;
      lo = nlo; hi = nhi;
      if (z == 0) { lo *= 0.125f; hi *= 0.125f; }  // 1/sqrt(64)
    }
    if (z == 1) {  // K stored transposed per head: KT[h][hd][s]
      KT[(size_t)(head * HDIM + hd_lo) * S_LEN + srow]        = (f16)lo;
      KT[(size_t)(head * HDIM + hd_lo + 32) * S_LEN + srow]   = (f16)hi;
    } else {       // Q / V stored [h][s][hd]
      f16* dst = (z == 0) ? Qh : Vh;
      dst[((size_t)head * S_LEN + srow) * HDIM + hd_lo]       = (f16)lo;
      dst[((size_t)head * S_LEN + srow) * HDIM + hd_lo + 32]  = (f16)hi;
    }
  }
}

// ---------------------------------------------------------------------------
// Kernel 2: causal flash attention, 1 wave per (16-row q-tile, head).
// Streams keys 32 at a time: 4 wmma QK^T + online softmax + 4 wmma PV.
// ---------------------------------------------------------------------------
__global__ __launch_bounds__(32) void attn_kernel(
    const f16* __restrict__ Qh, const f16* __restrict__ KT,
    const f16* __restrict__ Vh, f16* __restrict__ Aout) {
  const int qt = blockIdx.x;     // 0..255
  const int h  = blockIdx.y;     // 0..15
  const int lane = threadIdx.x & 31;
  const int l15 = lane & 15, lhalf = lane >> 4;

  __shared__ __align__(16) f16 Pt[16 * P_STRIDE];

  // Q fragments for both 32-wide K chunks of HD=64 (reused every step)
  V16U aq0, aq1;
  {
    const f16* qb = Qh + ((size_t)h * S_LEN + qt * 16 + l15) * HDIM;
    const int off = lhalf * 8;
    aq0.h[0] = *(const v8h*)(qb + off);
    aq0.h[1] = *(const v8h*)(qb + off + 16);
    aq1.h[0] = *(const v8h*)(qb + 32 + off);
    aq1.h[1] = *(const v8h*)(qb + 32 + off + 16);
  }

  float mrow[8], lrow[8];
#pragma unroll
  for (int v = 0; v < 8; ++v) { mrow[v] = -1e30f; lrow[v] = 0.0f; }
  v8f o0 = {}, o1 = {}, o2 = {}, o3 = {};

  const int qbase = qt * 16;
  const int jend  = qbase + 16;  // causal: j <= q < jend

  for (int j0 = 0; j0 < jend; j0 += 32) {
    // ---- scores: two 16-col tiles (j0.., j0+16..) ----
    v8f s0 = {}, s1 = {};
#pragma unroll
    for (int c = 0; c < 2; ++c) {
      V16U b0, b1;  // B frag from KT[h][hd][j]: lane col = j, elems = hd-k
#pragma unroll
      for (int s = 0; s < 16; ++s) {
        const f16* kb = KT + (size_t)(h * HDIM + c * 32 + lhalf * 8 + (s & 7) + (s >> 3) * 16) * S_LEN + j0 + l15;
        b0.v[s] = kb[0];
        b1.v[s] = kb[16];
      }
      const v16h a = c ? aq1.v : aq0.v;
      s0 = wmma_f16(a, b0.v, s0);
      s1 = wmma_f16(a, b1.v, s1);
    }

    // ---- causal mask + online softmax update ----
    float p0[8], p1[8];
#pragma unroll
    for (int v = 0; v < 8; ++v) {
      const int q = qbase + v + 8 * lhalf;
      float x0 = (j0 + l15 <= q)      ? s0[v] : -1e30f;
      float x1 = (j0 + 16 + l15 <= q) ? s1[v] : -1e30f;
      float mx = fmaxf(x0, x1);
#pragma unroll
      for (int d = 1; d < 16; d <<= 1) mx = fmaxf(mx, __shfl_xor(mx, d, 16));
      const float mnew = fmaxf(mrow[v], mx);
      const float sc = __expf(mrow[v] - mnew);
      const float e0 = __expf(x0 - mnew);
      const float e1 = __expf(x1 - mnew);
      float sum = e0 + e1;
#pragma unroll
      for (int d = 1; d < 16; d <<= 1) sum += __shfl_xor(sum, d, 16);
      lrow[v] = lrow[v] * sc + sum;
      mrow[v] = mnew;
      p0[v] = e0; p1[v] = e1;
      o0[v] *= sc; o1[v] *= sc; o2[v] *= sc; o3[v] *= sc;
    }

    // ---- transpose P (C layout -> A layout) through LDS ----
#pragma unroll
    for (int v = 0; v < 8; ++v) {
      const int row = v + 8 * lhalf;
      Pt[row * P_STRIDE + l15]      = (f16)p0[v];
      Pt[row * P_STRIDE + 16 + l15] = (f16)p1[v];
    }
    asm volatile("s_wait_dscnt 0x0" ::: "memory");  // DS in-order per wave
    V16U ap;
    {
      const int base = l15 * P_STRIDE + lhalf * 8;
      ap.h[0] = *(const v8h*)(&Pt[base]);
      ap.h[1] = *(const v8h*)(&Pt[base + 16]);
    }

    // ---- PV: B frag from Vh[h][j][hd]: lane col = hd, elems = j-k ----
    const f16* vb = Vh + ((size_t)h * S_LEN + j0 + lhalf * 8) * HDIM + l15;
    V16U bv0, bv1, bv2, bv3;
#pragma unroll
    for (int s = 0; s < 16; ++s) {
      const f16* vr = vb + ((s & 7) + (s >> 3) * 16) * HDIM;
      bv0.v[s] = vr[0]; bv1.v[s] = vr[16]; bv2.v[s] = vr[32]; bv3.v[s] = vr[48];
    }
    o0 = wmma_f16(ap.v, bv0.v, o0);
    o1 = wmma_f16(ap.v, bv1.v, o1);
    o2 = wmma_f16(ap.v, bv2.v, o2);
    o3 = wmma_f16(ap.v, bv3.v, o3);
  }

  // ---- finalize: divide by row sum, pack f16 into [S][D] for out-proj ----
#pragma unroll
  for (int v = 0; v < 8; ++v) {
    const int srow = qbase + v + 8 * lhalf;
    const float inv = 1.0f / lrow[v];
    f16* dst = Aout + (size_t)srow * DMODEL + h * HDIM;
    dst[l15 + 0]  = (f16)(o0[v] * inv);
    dst[l15 + 16] = (f16)(o1[v] * inv);
    dst[l15 + 32] = (f16)(o2[v] * inv);
    dst[l15 + 48] = (f16)(o3[v] * inv);
  }
}

// ---------------------------------------------------------------------------
// Kernel 3: output projection  Out(f32)[S,D] = Aout(f16)[S,D] @ Wo^T.
// Same async double-buffered WMMA GEMM; wave j owns cols [n0+j*32, +32).
// ---------------------------------------------------------------------------
__global__ __launch_bounds__(256) void oproj_kernel(
    const f16* __restrict__ A, const f16* __restrict__ WoT,
    float* __restrict__ Out) {
  __shared__ __align__(16) f16 Xt[2 * XT_BUF];
  __shared__ __align__(16) f16 Wt[2 * WT_BUF];

  const int m0 = blockIdx.x * 32;
  const int n0 = blockIdx.y * 128;
  const int tid = threadIdx.x;
  const int lane = tid & 31;
  const int w = tid >> 5;
  const int mi = w >> 2, j = w & 3;
  const int l15 = lane & 15, lhalf = lane >> 4;
  const int xr = tid >> 3, xc8 = (tid & 7) << 3;

  v8f acc0 = {}, acc1 = {};

  auto issue = [&](int kc, int buf) {
    async_ld_b128(A, (unsigned)(((m0 + xr) * DMODEL + kc + xc8) * 2),
                  &Xt[buf * XT_BUF + xr * XT_STRIDE + xc8]);
#pragma unroll
    for (int it = 0; it < 4; ++it) {
      const int idx = tid + it * 256;
      const int k = idx >> 4, c8 = (idx & 15) << 3;
      async_ld_b128(WoT, (unsigned)(((kc + k) * DMODEL + n0 + c8) * 2),
                    &Wt[buf * WT_BUF + k * WT_STRIDE + c8]);
    }
  };

  issue(0, 0);
  for (int st = 0; st < DMODEL / 64; ++st) {
    if (st + 1 < DMODEL / 64) {
      issue((st + 1) * 64, (st + 1) & 1);
      asm volatile("s_wait_asynccnt 0x5" ::: "memory");
    } else {
      asm volatile("s_wait_asynccnt 0x0" ::: "memory");
    }
    __syncthreads();

    const f16* xb = &Xt[(st & 1) * XT_BUF];
    const f16* wb = &Wt[(st & 1) * WT_BUF];
#pragma unroll
    for (int c = 0; c < 2; ++c) {
      V16U a;
      const int abase = (mi * 16 + l15) * XT_STRIDE + c * 32 + lhalf * 8;
      a.h[0] = *(const v8h*)(&xb[abase]);
      a.h[1] = *(const v8h*)(&xb[abase + 16]);

      V16U b0, b1;
      const int nb = j * 32 + l15;
#pragma unroll
      for (int s = 0; s < 16; ++s) {
        const int kb = (c * 32 + lhalf * 8 + (s & 7) + (s >> 3) * 16) * WT_STRIDE;
        b0.v[s] = wb[kb + nb];
        b1.v[s] = wb[kb + nb + 16];
      }

      acc0 = wmma_f16(a.v, b0.v, acc0);
      acc1 = wmma_f16(a.v, b1.v, acc1);
    }
    __syncthreads();
  }

#pragma unroll
  for (int v = 0; v < 8; ++v) {
    const int srow = m0 + mi * 16 + v + 8 * lhalf;
    const int col = n0 + j * 32 + l15;
    Out[(size_t)srow * DMODEL + col]      = acc0[v];
    Out[(size_t)srow * DMODEL + col + 16] = acc1[v];
  }
}

// ---------------------------------------------------------------------------
extern "C" void kernel_launch(void* const* d_in, const int* in_sizes, int n_in,
                              void* d_out, int out_size, void* d_ws,
                              size_t ws_size, hipStream_t stream) {
  (void)in_sizes; (void)n_in; (void)out_size; (void)ws_size;
  const float* X  = (const float*)d_in[0];  // [1,4096,1024]
  const float* TS = (const float*)d_in[1];  // [1,4096]
  const float* Wq = (const float*)d_in[2];
  const float* bq = (const float*)d_in[3];
  const float* Wk = (const float*)d_in[4];
  const float* bk = (const float*)d_in[5];
  const float* Wv = (const float*)d_in[6];
  const float* bv = (const float*)d_in[7];
  const float* Wo = (const float*)d_in[8];
  float* Out = (float*)d_out;               // [1,4096,1024] f32

  const size_t per = (size_t)NH * S_LEN * HDIM;  // 4,194,304 f16 each
  const size_t wsz = (size_t)DMODEL * DMODEL;    // 1,048,576 f16 each
  f16* Qh  = (f16*)d_ws;
  f16* KTw = Qh + per;
  f16* Vh  = KTw + per;
  f16* Ao  = Vh + per;   // [S][D]
  f16* Xf  = Ao + per;   // [S][D]
  f16* WqT = Xf + per;
  f16* WkT = WqT + wsz;
  f16* WvT = WkT + wsz;
  f16* WoT = WvT + wsz;

  convert_x_kernel<<<(S_LEN * DMODEL) / (256 * 8), 256, 0, stream>>>(X, Xf);
  transpose_w_kernel<<<dim3(DMODEL / 32, DMODEL / 32, 4), 256, 0, stream>>>(
      Wq, Wk, Wv, Wo, WqT, WkT, WvT, WoT);
  qkv_rope_kernel<<<dim3(S_LEN / 32, DMODEL / 128, 3), 256, 0, stream>>>(
      Xf, TS, WqT, bq, WkT, bk, WvT, bv, Qh, KTw, Vh);
  attn_kernel<<<dim3(S_LEN / 16, NH), 32, 0, stream>>>(Qh, KTw, Vh, Ao);
  oproj_kernel<<<dim3(S_LEN / 32, DMODEL / 128), 256, 0, stream>>>(Ao, WoT, Out);
}
